// MPNEncoder_62242666054064
// MI455X (gfx1250) — matrix operand order repaired and checked by reading, not compiled, not run on hardware
//
#include <hip/hip_runtime.h>

// Problem constants (match reference)
#define N_ATOMS 50000
#define N_BONDS 100000
#define MAX_NB  6
#define F_ATOMC 133
#define F_BONDC 147
#define HID     600
#define DEPTH   6

// Block = 256 threads = 8 wave32. Block owns 64(M) rows x ALL 600 cols.
// A strip (64 x K, bf16) staged to LDS once; W tiles streamed through a
// double-buffered 128-col LDS tile; 5 internal N-blocks reuse the A strip.
#define BM 64
#define BN 128
#define BK 32
#define NBLK ((HID + BN - 1) / BN)   // 5
#define LDSK 40   // W-tile K-stride: 80B rows, 16B-aligned chunks, conflict-free

typedef __attribute__((ext_vector_type(16))) __bf16 v16bf;
typedef __attribute__((ext_vector_type(8)))  __bf16 v8bf;
typedef __attribute__((ext_vector_type(4)))  __bf16 v4bf;
typedef __attribute__((ext_vector_type(8)))  float  v8f;

__device__ __forceinline__ int imin(int a, int b) { return a < b ? a : b; }

// ---------------- full-K A-strip staging (once per block) ------------------
// Stage BM x KP32 (K rounded up to 32) bf16 elements into Asl.
// M out-of-range rows CLAMPED (outputs masked in epilogue); k >= K zero-filled.
template<int AMODE, int KPAD>
__device__ __forceinline__ void stage_a_strip(
    __bf16 (*Asl)[KPAD], int tid, int blockM, int M, int K, int KP32,
    const float* __restrict__ A0, const float* __restrict__ A1,
    const float* __restrict__ A2, const int* __restrict__ idx0,
    const int* __restrict__ idx1)
{
    const int kq = KP32 >> 2;                   // 4-wide chunks per row
    for (int i = tid; i < BM * kq; i += 256) {
        const int row  = i / kq;
        const int k4   = (i - row * kq) * 4;
        const int grow = imin(blockM + row, M - 1);
        float a0, a1, a2, a3;
        if (AMODE == 1) {
            const int sa = idx0[grow];
            const int rb = idx1[grow];
            if (k4 + 4 <= K) {
                const float4 va = *(const float4*)(A1 + (size_t)sa * HID + k4);
                const float4 vb = *(const float4*)(A2 + (size_t)rb * HID + k4);
                a0 = va.x - vb.x; a1 = va.y - vb.y;
                a2 = va.z - vb.z; a3 = va.w - vb.w;
            } else {
                a0 = (k4+0 < K) ? A1[(size_t)sa*HID + k4+0] - A2[(size_t)rb*HID + k4+0] : 0.f;
                a1 = (k4+1 < K) ? A1[(size_t)sa*HID + k4+1] - A2[(size_t)rb*HID + k4+1] : 0.f;
                a2 = (k4+2 < K) ? A1[(size_t)sa*HID + k4+2] - A2[(size_t)rb*HID + k4+2] : 0.f;
                a3 = (k4+3 < K) ? A1[(size_t)sa*HID + k4+3] - A2[(size_t)rb*HID + k4+3] : 0.f;
            }
        } else if (AMODE == 0) {
            const float* p = A0 + (size_t)grow * K + k4;
            a0 = (k4+0 < K) ? p[0] : 0.f;
            a1 = (k4+1 < K) ? p[1] : 0.f;
            a2 = (k4+2 < K) ? p[2] : 0.f;
            a3 = (k4+3 < K) ? p[3] : 0.f;
        } else {   // AMODE 2: concat [f_atoms(133) | a_message(600)]
            float v[4];
            #pragma unroll
            for (int t = 0; t < 4; ++t) {
                const int gk = k4 + t;
                const float* p = (gk < F_ATOMC)
                    ? (A0 + (size_t)grow * F_ATOMC + gk)
                    : (A1 + (size_t)grow * HID + (gk - F_ATOMC));
                v[t] = (gk < K) ? *p : 0.f;
            }
            a0 = v[0]; a1 = v[1]; a2 = v[2]; a3 = v[3];
        }
        v4bf t;
        t[0] = (__bf16)a0; t[1] = (__bf16)a1;
        t[2] = (__bf16)a2; t[3] = (__bf16)a3;
        *(v4bf*)&Asl[row][k4] = t;
    }
}

// ---------------- W-tile staging (BK x BN, double-buffered) ----------------
// 16 per thread, 4 chunks of 4-along-K; coalesced b32 loads.
// N out-of-range cols CLAMPED (outputs masked); K remainder zero-filled.
template<bool KG>
__device__ __forceinline__ void load_b_tile(
    float b[16], int tid, int blockN, int k0, int K, const float* __restrict__ W)
{
    #pragma unroll
    for (int j = 0; j < 4; ++j) {
        const int c  = tid + j * 256;
        const int n  = c & 127;
        const int k4 = (c >> 7) * 4;
        const int gn = imin(blockN + n, HID - 1);
        #pragma unroll
        for (int t = 0; t < 4; ++t) {
            const int gk = k0 + k4 + t;
            b[j*4+t] = (!KG || gk < K) ? W[(size_t)gk * HID + gn] : 0.f;
        }
    }
}

__device__ __forceinline__ void store_b_tile(
    const float b[16], int tid, __bf16 (*Bs)[LDSK])
{
    #pragma unroll
    for (int j = 0; j < 4; ++j) {
        const int c  = tid + j * 256;
        const int n  = c & 127;
        const int k4 = (c >> 7) * 4;
        v4bf t;
        t[0] = (__bf16)b[j*4+0]; t[1] = (__bf16)b[j*4+1];
        t[2] = (__bf16)b[j*4+2]; t[3] = (__bf16)b[j*4+3];
        *(v4bf*)&Bs[n][k4] = t;
    }
}

// ---------------- GEMM kernel ----------------------------------------------
// AMODE: 0 = A0[row*K+k]
//        1 = A1[idx0[row]*HID+k] - A2[idx1[row]*HID+k]   (directed-bond update)
//        2 = k<F_ATOMC ? f_atoms : a_message             (concat)
// EPI:   0 = out0 = acc (inp), out1 = relu(acc)
//        1 = out0 = relu(bias[row*HID+col] + acc)        (bias = inp)
//        2 = out0 = relu(acc + bias[col])                (bias = b_o)
// KPAD: LDS K-stride for the A strip (>= ceil32(K); 2*KPAD % 16 == 0 and
//       (KPAD/2) mod 64 has gcd 4 with banks -> conflict-free frag reads).
template<int AMODE, int EPI, int KPAD>
__global__ __launch_bounds__(256, 2)
void gemm_wmma_kernel(const float* __restrict__ A0,
                      const float* __restrict__ A1,
                      const float* __restrict__ A2,
                      const int*   __restrict__ idx0,
                      const int*   __restrict__ idx1,
                      const float* __restrict__ W,
                      const float* __restrict__ bias,
                      float* __restrict__ out0,
                      float* __restrict__ out1,
                      int M, int K)
{
    __shared__ __align__(16) __bf16 Asl[BM][KPAD];
    __shared__ __align__(16) __bf16 Bs[2][BN][LDSK];

    const int tid  = threadIdx.x;
    const int lane = tid & 31;
    const int wave = tid >> 5;
    const int wm   = wave & 3;
    const int wn   = wave >> 2;
    const int l15  = lane & 15;
    const int half = lane >> 4;

    const int blockM = blockIdx.x * BM;
    const int nsteps = (K + BK - 1) / BK;
    const int KP32   = nsteps * BK;

    // Stage the 64 x K A strip once (bf16); reused by all 5 N-blocks.
    stage_a_strip<AMODE, KPAD>(Asl, tid, blockM, M, K, KP32, A0, A1, A2, idx0, idx1);

    float breg[16];

    for (int nb = 0; nb < NBLK; ++nb) {
        const int blockN = nb * BN;
        __syncthreads();   // A strip ready (nb==0) / prior Bs reads done

        // prologue: W tile for k0=0 into buffer 0
        if (BK > K) load_b_tile<true >(breg, tid, blockN, 0, K, W);
        else        load_b_tile<false>(breg, tid, blockN, 0, K, W);
        store_b_tile(breg, tid, Bs[0]);
        __syncthreads();

        v8f acc[4] = {};

        for (int s = 0; s < nsteps; ++s) {
            const int buf = s & 1;
            const bool have_next = (s + 1) < nsteps;

            // issue next W tile's global loads before compute
            if (have_next) {
                const int k0 = (s + 1) * BK;
                if (k0 + BK > K) load_b_tile<true >(breg, tid, blockN, k0, K, W);
                else             load_b_tile<false>(breg, tid, blockN, k0, K, W);
            }

            // ---- fragments then WMMAs ----
            v16bf afrag;
            {
                const int m  = wm * 16 + l15;
                const int kb = s * BK;
                union { v16bf v; v8bf h[2]; } u;
                u.h[0] = *(const v8bf*)&Asl[m][kb + half * 8];
                u.h[1] = *(const v8bf*)&Asl[m][kb + 16 + half * 8];
                afrag = u.v;
            }
            v16bf bfrag[4];
            #pragma unroll
            for (int t = 0; t < 4; ++t) {
                const int n = wn * 64 + t * 16 + l15;
                union { v16bf v; v8bf h[2]; } u;
                u.h[0] = *(const v8bf*)&Bs[buf][n][half * 8];
                u.h[1] = *(const v8bf*)&Bs[buf][n][16 + half * 8];
                bfrag[t] = u.v;
            }
            #pragma unroll
            for (int t = 0; t < 4; ++t) {
                acc[t] = __builtin_amdgcn_wmma_f32_16x16x32_bf16(
                    false, afrag, false, bfrag[t], (short)0, acc[t], false, false);
            }

            if (have_next) {
                store_b_tile(breg, tid, Bs[buf ^ 1]);
                __syncthreads();
            }
        }

        // ---- epilogue for this N-block (C/D: M = r + 8*half, N = l15) ----
        #pragma unroll
        for (int t = 0; t < 4; ++t) {
            #pragma unroll
            for (int r = 0; r < 8; ++r) {
                const int grow = blockM + wm * 16 + r + 8 * half;
                const int gcol = blockN + wn * 64 + t * 16 + l15;
                if (grow < M && gcol < HID) {
                    const float v = acc[t][r];
                    const size_t o = (size_t)grow * HID + gcol;
                    if (EPI == 0) {
                        out0[o] = v;                      // inp (pre-activation)
                        out1[o] = v > 0.f ? v : 0.f;      // message = relu(inp)
                    } else if (EPI == 1) {
                        const float s2 = bias[o] + v;     // relu(inp + upd @ W_h)
                        out0[o] = s2 > 0.f ? s2 : 0.f;
                    } else {
                        const float s2 = v + bias[gcol];  // relu(acc + b_o)
                        out0[o] = s2 > 0.f ? s2 : 0.f;
                    }
                }
            }
        }
    }
}

// a_message[a, :] = sum_j message[a2b[a, j], :]  — float4-vectorized gather-sum
__global__ __launch_bounds__(256)
void aggregate_kernel(const float* __restrict__ message,
                      const int*   __restrict__ a2b,
                      float* __restrict__ amsg)
{
    const int C4 = HID / 4; // 150
    const size_t i = (size_t)blockIdx.x * blockDim.x + threadIdx.x;
    if (i >= (size_t)N_ATOMS * C4) return;
    const int a  = (int)(i / C4);
    const int c4 = (int)(i % C4);
    const int* nb = a2b + (size_t)a * MAX_NB;
    float4 s = {0.f, 0.f, 0.f, 0.f};
    #pragma unroll
    for (int j = 0; j < MAX_NB; ++j) {
        const float4 v = *(const float4*)(message + (size_t)nb[j] * HID + c4 * 4);
        s.x += v.x; s.y += v.y; s.z += v.z; s.w += v.w;
    }
    *(float4*)(amsg + (size_t)a * HID + c4 * 4) = s;
}

extern "C" void kernel_launch(void* const* d_in, const int* in_sizes, int n_in,
                              void* d_out, int out_size, void* d_ws, size_t ws_size,
                              hipStream_t stream)
{
    const float* f_atoms = (const float*)d_in[0];
    const float* f_bonds = (const float*)d_in[1];
    const int*   a2b     = (const int*)d_in[2];
    const int*   b2a     = (const int*)d_in[3];
    const int*   b2revb  = (const int*)d_in[4];
    const float* W_i     = (const float*)d_in[5];
    const float* W_h     = (const float*)d_in[6];
    const float* W_o     = (const float*)d_in[7];
    const float* b_o     = (const float*)d_in[8];
    float* out = (float*)d_out;

    (void)in_sizes; (void)n_in; (void)out_size; (void)ws_size;

    char* ws = (char*)d_ws;
    const size_t bondBytes = (size_t)N_BONDS * HID * sizeof(float); // 240 MB
    float* inp  = (float*)(ws);
    float* msg0 = (float*)(ws + bondBytes);
    float* msg1 = (float*)(ws + 2 * bondBytes);
    float* amsg = (float*)(ws + 3 * bondBytes);

    dim3 blk(256);
    dim3 grdB((N_BONDS + BM - 1) / BM);
    dim3 grdA((N_ATOMS + BM - 1) / BM);
    dim3 aggGrid((unsigned)(((size_t)N_ATOMS * (HID / 4) + 255) / 256));

    // inp = f_bonds @ W_i ; message = relu(inp)   (K=147 -> KPAD=168)
    gemm_wmma_kernel<0, 0, 168><<<grdB, blk, 0, stream>>>(
        f_bonds, nullptr, nullptr, nullptr, nullptr,
        W_i, nullptr, inp, msg0, N_BONDS, F_BONDC);

    float* cur = msg0;
    float* nxt = msg1;
    for (int it = 0; it < DEPTH - 1; ++it) {
        aggregate_kernel<<<aggGrid, blk, 0, stream>>>(cur, a2b, amsg);
        // message = relu(inp + (amsg[b2a] - message[b2revb]) @ W_h)  (K=600 -> KPAD=616)
        gemm_wmma_kernel<1, 1, 616><<<grdB, blk, 0, stream>>>(
            nullptr, amsg, cur, b2a, b2revb,
            W_h, inp, nxt, nullptr, N_BONDS, HID);
        float* t = cur; cur = nxt; nxt = t;
    }

    aggregate_kernel<<<aggGrid, blk, 0, stream>>>(cur, a2b, amsg);
    // out = relu([f_atoms | amsg] @ W_o + b_o)    (K=733 -> KPAD=744)
    gemm_wmma_kernel<2, 2, 744><<<grdA, blk, 0, stream>>>(
        f_atoms, amsg, nullptr, nullptr, nullptr,
        W_o, b_o, out, nullptr, N_ATOMS, F_ATOMC + HID);
}
